// Graph_GAN_29832842838350
// MI455X (gfx1250) — matrix operand order repaired and checked by reading, not compiled, not run on hardware
//
#include <hip/hip_runtime.h>
#include <hip/hip_bf16.h>
#include <math.h>

// ---------------------------------------------------------------------------
// Graph-GAN forward: edge MLP over all pairs (WMMA f16 -> f32), sum over j,
// node MLP, tanh of first 3 features.  MI455X / gfx1250, wave32 WMMA.
// ---------------------------------------------------------------------------

typedef __attribute__((ext_vector_type(16))) _Float16 v16h;
typedef __attribute__((ext_vector_type(8)))  _Float16 v8h;
typedef __attribute__((ext_vector_type(8)))  float    v8f;

#define BATCH 64
#define NNODE 100
#define HDIM  32
#define NTI   7          // ceil(100/16) sender tiles
#define ALPHA 0.2f

__device__ __forceinline__ v8f wmma_f16(v16h a, v16h b, v8f c) {
    // D = A(16x32 f16) * B(32x16 f16) + C(16x16 f32)
    return __builtin_amdgcn_wmma_f32_16x16x32_f16(
        /*neg_a=*/false, a, /*neg_b=*/false, b,
        /*c_mod=*/(short)0, c, /*reuse_a=*/false, /*reuse_b=*/false);
}

// A-operand (16x32 f16 tile, k-tile kt) from a row-major f16 matrix in LDS,
// ld = row stride in halves (multiple of 8).
// Layout: lane m=l&15; lanes 0-15 chunk K = kt*32 + {0..15},
//                      lanes 16-31 chunk K = kt*32 + {16..31}.
__device__ __forceinline__ v16h load_a_lds(const _Float16* base, int ld, int kt) {
    const int lane = threadIdx.x & 31;
    const int m    = lane & 15;
    const int kh   = lane >> 4;
    const _Float16* p = base + m * ld + kt * 32 + kh * 16;
    v8h lo, hi;
    __builtin_memcpy(&lo, p,      16);   // ds_load_b128
    __builtin_memcpy(&hi, p + 8, 16);    // ds_load_b128
    v16h a;
#pragma unroll
    for (int t = 0; t < 8; ++t) { a[t] = lo[t]; a[t + 8] = hi[t]; }
    return a;
}

// B-operand (32x16 f16 tile) from padded f16 weight W[out_f][kp] row-major,
// B[k][n] = W[nt*16+n][kt*32+k].  Per lane: 16 consecutive K halves (32 B).
__device__ __forceinline__ v16h load_b_w(const _Float16* w, int kp, int nt, int kt) {
    const int lane = threadIdx.x & 31;
    const int n    = lane & 15;
    const int kh   = lane >> 4;
    const _Float16* p = w + (size_t)(nt * 16 + n) * kp + kt * 32 + kh * 16;
    v16h b;
    __builtin_memcpy(&b, p, 32);         // 2x global_load_b128
    return b;
}

// C initialized with per-column bias (f32).
__device__ __forceinline__ v8f bias_c(const float* __restrict__ bias, int nt) {
    const float bv = bias[nt * 16 + (threadIdx.x & 15)];
    v8f c;
#pragma unroll
    for (int v = 0; v < 8; ++v) c[v] = bv;
    return c;
}

// Store a 16x16 f32 C tile into row-major f16 LDS activation (with leaky_relu).
// C layout: lane n = nt*16 + (l&15);  row m = v + 8*(l>>4).
__device__ __forceinline__ void store_c_lds(_Float16* base, int ld, int nt, v8f c) {
    const int lane = threadIdx.x & 31;
    const int n    = nt * 16 + (lane & 15);
    const int mh   = (lane >> 4) * 8;
#pragma unroll
    for (int v = 0; v < 8; ++v) {
        float y = c[v];
        y = fmaxf(y, ALPHA * y);                 // leaky_relu, alpha in (0,1)
        base[(mh + v) * ld + n] = (_Float16)y;   // ds_store_b16
    }
}

// ---------------------------------------------------------------------------
// Weight f32 -> f16 with K zero-padding (kin -> kp).
// ---------------------------------------------------------------------------
__global__ void convert_w_kernel(const float* __restrict__ src, _Float16* __restrict__ dst,
                                 int out_f, int kin, int kp) {
    int idx = blockIdx.x * blockDim.x + threadIdx.x;
    if (idx >= out_f * kp) return;
    int o = idx / kp, k = idx % kp;
    dst[idx] = (k < kin) ? (_Float16)src[o * kin + k] : (_Float16)0.0f;
}

// ---------------------------------------------------------------------------
// Kernel 1: edge MLP over all (i,j) pairs + sum over j.
// Block = (b, i-tile of 16 senders), 4 waves; wave w handles j = w, w+4, ...
// ---------------------------------------------------------------------------
__global__ __launch_bounds__(128) void edge_kernel(
    const float* __restrict__ x,
    const _Float16* __restrict__ W0, const float* __restrict__ b0,   // 96 x 96(pad65)
    const _Float16* __restrict__ W1, const float* __restrict__ b1,   // 160 x 96
    const _Float16* __restrict__ W2, const float* __restrict__ b2,   // 192 x 160
    float* __restrict__ asum)                                        // [B][N][192]
{
    __shared__ _Float16 feat_s[4][16 * 96];
    __shared__ _Float16 act1_s[4][16 * 96];
    __shared__ _Float16 act2_s[4][16 * 160];
    __shared__ float    ssum[16 * 192];

    const int tid  = threadIdx.x;
    const int lane = tid & 31;
    const int wave = tid >> 5;
    const int b    = blockIdx.x / NTI;
    const int it   = blockIdx.x % NTI;
    const int i0   = it * 16;

    for (int idx = tid; idx < 16 * 192; idx += 128) ssum[idx] = 0.0f;

    _Float16* F  = feat_s[wave];
    _Float16* A1 = act1_s[wave];
    _Float16* A2 = act2_s[wave];

    // Static part of the edge feature tile: cols [0,32) = x_i, cols [65,96) = 0.
    for (int idx = lane; idx < 16 * 96; idx += 32) {
        int m = idx / 96, k = idx % 96;
        int i = i0 + m;
        float v = 0.0f;
        if (k < 32 && i < NNODE) v = x[((size_t)b * NNODE + i) * HDIM + k];
        F[m * 96 + k] = (_Float16)v;
    }
    __builtin_amdgcn_wave_barrier();

    v8f sum[12];
#pragma unroll
    for (int t = 0; t < 12; ++t)
#pragma unroll
        for (int v = 0; v < 8; ++v) sum[t][v] = 0.0f;

    __syncthreads();   // ssum zeroed before any wave's atomics

    for (int j = wave; j < NNODE; j += 4) {
        const float* xj = x + ((size_t)b * NNODE + j) * HDIM;

        // Stage x_j into cols [32,64): lane l -> row m=l/2, 16 halves.
        {
            int m  = lane >> 1;
            int c0 = (lane & 1) * 16;
            v8h p0, p1;
#pragma unroll
            for (int t = 0; t < 8; ++t) p0[t] = (_Float16)xj[c0 + t];
#pragma unroll
            for (int t = 0; t < 8; ++t) p1[t] = (_Float16)xj[c0 + 8 + t];
            __builtin_memcpy(&F[m * 96 + 32 + c0], &p0, 16);
            __builtin_memcpy(&F[m * 96 + 40 + c0], &p1, 16);
        }
        // dist column (col 64): lanes 0..15 handle row m = lane.
        if (lane < 16) {
            int i = i0 + lane;
            float xi0 = 0.0f, xi1 = 0.0f;
            if (i < NNODE) {
                xi0 = x[((size_t)b * NNODE + i) * HDIM + 0];
                xi1 = x[((size_t)b * NNODE + i) * HDIM + 1];
            }
            float d0 = xj[0] - xi0 + 1e-12f;
            float d1 = xj[1] - xi1 + 1e-12f;
            F[lane * 96 + 64] = (_Float16)sqrtf(d0 * d0 + d1 * d1);
        }
        __builtin_amdgcn_wave_barrier();

        // Layer 1: [16x96(K=65 padded)] -> [16x96]
        for (int nt = 0; nt < 6; ++nt) {
            v8f c = bias_c(b0, nt);
#pragma unroll
            for (int kt = 0; kt < 3; ++kt)
                c = wmma_f16(load_a_lds(F, 96, kt), load_b_w(W0, 96, nt, kt), c);
            store_c_lds(A1, 96, nt, c);
        }
        __builtin_amdgcn_wave_barrier();

        // Layer 2: [16x96] -> [16x160]
        for (int nt = 0; nt < 10; ++nt) {
            v8f c = bias_c(b1, nt);
#pragma unroll
            for (int kt = 0; kt < 3; ++kt)
                c = wmma_f16(load_a_lds(A1, 96, kt), load_b_w(W1, 96, nt, kt), c);
            store_c_lds(A2, 160, nt, c);
        }
        __builtin_amdgcn_wave_barrier();

        // Layer 3: [16x160] -> [16x192], leaky_relu, accumulate over j in f32.
        for (int nt = 0; nt < 12; ++nt) {
            v8f c = bias_c(b2, nt);
#pragma unroll
            for (int kt = 0; kt < 5; ++kt)
                c = wmma_f16(load_a_lds(A2, 160, kt), load_b_w(W2, 160, nt, kt), c);
#pragma unroll
            for (int v = 0; v < 8; ++v) {
                float y = c[v];
                sum[nt][v] += fmaxf(y, ALPHA * y);
            }
        }
        __builtin_amdgcn_wave_barrier();
    }

    // Cross-wave reduction: LDS f32 atomics into shared tile.
    for (int nt = 0; nt < 12; ++nt) {
        const int n  = nt * 16 + (lane & 15);
        const int mh = (lane >> 4) * 8;
#pragma unroll
        for (int v = 0; v < 8; ++v)
            atomicAdd(&ssum[(mh + v) * 192 + n], sum[nt][v]);   // ds_add_f32
    }
    __syncthreads();

    for (int idx = tid; idx < 16 * 192; idx += 128) {
        int m = idx / 192, f = idx % 192;
        int i = i0 + m;
        if (i < NNODE) asum[((size_t)b * NNODE + i) * 192 + f] = ssum[idx];
    }
}

// ---------------------------------------------------------------------------
// Kernel 2: node MLP 224 -> 256 -> 256 -> 32, tanh of first 3 outputs.
// Block = 1 wave handling a 16-node tile.
// ---------------------------------------------------------------------------
__global__ __launch_bounds__(32) void node_kernel(
    const float* __restrict__ x,
    const float* __restrict__ asum,
    const _Float16* __restrict__ Wn0, const float* __restrict__ bn0,  // 256 x 224
    const _Float16* __restrict__ Wn1, const float* __restrict__ bn1,  // 256 x 256
    const _Float16* __restrict__ Wn2, const float* __restrict__ bn2,  // 32  x 256
    float* __restrict__ out)                                          // [B][N][3]
{
    __shared__ _Float16 feat[16 * 224];
    __shared__ _Float16 a1[16 * 256];
    __shared__ _Float16 a2[16 * 256];

    const int lane = threadIdx.x & 31;
    const int b    = blockIdx.x / NTI;
    const int it   = blockIdx.x % NTI;
    const int i0   = it * 16;

    // Stage concat([A_sum, x]) as f16.
    for (int idx = lane; idx < 16 * 224; idx += 32) {
        int m = idx / 224, f = idx % 224;
        int i = i0 + m;
        float v = 0.0f;
        if (i < NNODE)
            v = (f < 192) ? asum[((size_t)b * NNODE + i) * 192 + f]
                          : x[((size_t)b * NNODE + i) * HDIM + (f - 192)];
        feat[idx] = (_Float16)v;
    }
    __builtin_amdgcn_wave_barrier();

    for (int nt = 0; nt < 16; ++nt) {          // 224 -> 256
        v8f c = bias_c(bn0, nt);
#pragma unroll
        for (int kt = 0; kt < 7; ++kt)
            c = wmma_f16(load_a_lds(feat, 224, kt), load_b_w(Wn0, 224, nt, kt), c);
        store_c_lds(a1, 256, nt, c);
    }
    __builtin_amdgcn_wave_barrier();

    for (int nt = 0; nt < 16; ++nt) {          // 256 -> 256
        v8f c = bias_c(bn1, nt);
#pragma unroll
        for (int kt = 0; kt < 8; ++kt)
            c = wmma_f16(load_a_lds(a1, 256, kt), load_b_w(Wn1, 256, nt, kt), c);
        store_c_lds(a2, 256, nt, c);
    }
    __builtin_amdgcn_wave_barrier();

    // Final layer: only columns 0..2 survive the tanh slice -> one N-tile.
    {
        v8f c = bias_c(bn2, 0);
#pragma unroll
        for (int kt = 0; kt < 8; ++kt)
            c = wmma_f16(load_a_lds(a2, 256, kt), load_b_w(Wn2, 256, 0, kt), c);
        const int n  = lane & 15;
        const int mh = (lane >> 4) * 8;
        if (n < 3) {
#pragma unroll
            for (int v = 0; v < 8; ++v) {
                int i = i0 + mh + v;
                if (i < NNODE)
                    out[((size_t)b * NNODE + i) * 3 + n] = tanhf(c[v]);
            }
        }
    }
}

// ---------------------------------------------------------------------------
// Host launch
// ---------------------------------------------------------------------------
extern "C" void kernel_launch(void* const* d_in, const int* in_sizes, int n_in,
                              void* d_out, int out_size, void* d_ws, size_t ws_size,
                              hipStream_t stream) {
    (void)in_sizes; (void)n_in; (void)out_size; (void)ws_size;

    const float* x     = (const float*)d_in[0];
    const float* fe_w0 = (const float*)d_in[1];
    const float* fe_b0 = (const float*)d_in[2];
    const float* fe_w1 = (const float*)d_in[3];
    const float* fe_b1 = (const float*)d_in[4];
    const float* fe_w2 = (const float*)d_in[5];
    const float* fe_b2 = (const float*)d_in[6];
    const float* fn_w0 = (const float*)d_in[7];
    const float* fn_b0 = (const float*)d_in[8];
    const float* fn_w1 = (const float*)d_in[9];
    const float* fn_b1 = (const float*)d_in[10];
    const float* fn_w2 = (const float*)d_in[11];
    const float* fn_b2 = (const float*)d_in[12];
    float* out = (float*)d_out;

    // Workspace layout: f32 A_sum then padded f16 weights.
    char* ws = (char*)d_ws;
    float* asum = (float*)ws;                                // 64*100*192 f32
    size_t off = (size_t)BATCH * NNODE * 192 * sizeof(float);
    _Float16* W0  = (_Float16*)(ws + off); off += (size_t) 96 *  96 * 2;
    _Float16* W1  = (_Float16*)(ws + off); off += (size_t)160 *  96 * 2;
    _Float16* W2  = (_Float16*)(ws + off); off += (size_t)192 * 160 * 2;
    _Float16* Wn0 = (_Float16*)(ws + off); off += (size_t)256 * 224 * 2;
    _Float16* Wn1 = (_Float16*)(ws + off); off += (size_t)256 * 256 * 2;
    _Float16* Wn2 = (_Float16*)(ws + off); off += (size_t) 32 * 256 * 2;

    auto conv = [&](const float* src, _Float16* dst, int of, int kin, int kp) {
        int n = of * kp;
        convert_w_kernel<<<(n + 255) / 256, 256, 0, stream>>>(src, dst, of, kin, kp);
    };
    conv(fe_w0, W0,   96,  65,  96);
    conv(fe_w1, W1,  160,  96,  96);
    conv(fe_w2, W2,  192, 160, 160);
    conv(fn_w0, Wn0, 256, 224, 224);
    conv(fn_w1, Wn1, 256, 256, 256);
    conv(fn_w2, Wn2,  32, 256, 256);

    edge_kernel<<<BATCH * NTI, 128, 0, stream>>>(x, W0, fe_b0, W1, fe_b1, W2, fe_b2, asum);
    node_kernel<<<BATCH * NTI, 32, 0, stream>>>(x, asum, Wn0, fn_b0, Wn1, fn_b1, Wn2, fn_b2, out);
}
